// SelfAttention_88716844466579
// MI455X (gfx1250) — compile-verified, matrix operand order checked by reading, not run on hardware
//
#include <hip/hip_runtime.h>

// ---------------------------------------------------------------------------
// SelfAttention (faithful-to-source variant), gfx1250 / CDNA5 / wave32.
// Pipeline:
//   0) pack fp32 -> bf16 for x, Wq, Wk, Wv (one-time, bandwidth-trivial)
//   1) QKV GEMM: bf16 v_wmma_f32_16x16x32_bf16, fp32 accumulate, fused bias;
//      Q,K stored bf16 (WMMA-ready), V stored fp32.
//   2) per-token 32x32 scores via 2x2 WMMA tiles; LDS transpose so each lane
//      owns one score row; register softmax; elementwise multiply with V.
// ---------------------------------------------------------------------------

typedef __attribute__((ext_vector_type(16))) __bf16 bf16x16;
typedef __attribute__((ext_vector_type(8)))  __bf16 bf16x8;
typedef __attribute__((ext_vector_type(8)))  float  f32x8;
typedef __attribute__((ext_vector_type(4)))  float  f32x4;

// 16x32 bf16 WMMA A/B fragment from a bf16 row-major matrix.
// Caller passes p = &M[row + (lane&15)][k0 + 8*(lane>>4)].
// Per-lane layout (ISA 7.12.2): elements 0..7 = K+0..7, 8..15 = K+16..23.
static __device__ __forceinline__ bf16x16 load_frag_bf16(const __bf16* __restrict__ p) {
    bf16x8 lo = *(const bf16x8*)(p);        // one global_load_b128
    bf16x8 hi = *(const bf16x8*)(p + 16);   // one global_load_b128
    bf16x16 f;
#pragma unroll
    for (int i = 0; i < 8; ++i) { f[i] = lo[i]; f[8 + i] = hi[i]; }
    return f;
}

// ---------------------------------------------------------------------------
// Kernel 0: fp32 -> bf16 pack (n must be a multiple of 8; grid sized exactly)
// ---------------------------------------------------------------------------
__global__ __launch_bounds__(256) void pack_bf16_kernel(
    const float* __restrict__ src, __bf16* __restrict__ dst, size_t n)
{
    size_t i = ((size_t)blockIdx.x * blockDim.x + threadIdx.x) * 8;
    const size_t stride = (size_t)gridDim.x * blockDim.x * 8;
    for (; i + 7 < n; i += stride) {
        f32x4 a = *(const f32x4*)(src + i);
        f32x4 b = *(const f32x4*)(src + i + 4);
        bf16x8 o;
#pragma unroll
        for (int j = 0; j < 4; ++j) { o[j] = (__bf16)a[j]; o[4 + j] = (__bf16)b[j]; }
        *(bf16x8*)(dst + i) = o;
    }
}

// ---------------------------------------------------------------------------
// Kernel 1: C[m,n] = sum_k X[m,k] * W[n,k] + bias[n]   (x @ W^T + b), bf16 in.
// 8 waves = 2(M) x 4(N); wave tile 32x64; WG tile 64x256; K-step 32.
// Hot loop per wave: 12 global_load_b128 + 8 v_wmma.
// ---------------------------------------------------------------------------
template <bool STORE_BF16>
__global__ __launch_bounds__(256) void qkv_gemm_kernel(
    const __bf16* __restrict__ X, const __bf16* __restrict__ W,
    const float* __restrict__ bias, void* __restrict__ Out,
    int M, int K, int N)
{
    const int lane     = threadIdx.x & 31;
    const int wave     = threadIdx.x >> 5;
    const int mw       = wave >> 2;              // 0..1
    const int nw       = wave & 3;               // 0..3
    const int mBase    = blockIdx.x * 64 + mw * 32;
    const int nBase    = blockIdx.y * 256 + nw * 64;
    const int laneRow  = lane & 15;
    const int laneKOff = (lane >> 4) << 3;       // 0 or 8

    f32x8 acc[2][4] = {};

    const __bf16* xRow0 = X + (size_t)(mBase + laneRow) * K + laneKOff;
    const __bf16* xRow1 = X + (size_t)(mBase + 16 + laneRow) * K + laneKOff;
    const __bf16* wRow[4];
#pragma unroll
    for (int ni = 0; ni < 4; ++ni)
        wRow[ni] = W + (size_t)(nBase + ni * 16 + laneRow) * K + laneKOff;

    for (int k0 = 0; k0 < K; k0 += 32) {
        if (k0 + 32 < K) {
            __builtin_prefetch(xRow0 + k0 + 32, 0, 0);
            __builtin_prefetch(wRow[0] + k0 + 32, 0, 0);
        }
        bf16x16 aF[2];
        aF[0] = load_frag_bf16(xRow0 + k0);
        aF[1] = load_frag_bf16(xRow1 + k0);
        bf16x16 bF[4];
#pragma unroll
        for (int ni = 0; ni < 4; ++ni)
            bF[ni] = load_frag_bf16(wRow[ni] + k0);

#pragma unroll
        for (int mi = 0; mi < 2; ++mi)
#pragma unroll
            for (int ni = 0; ni < 4; ++ni)
                acc[mi][ni] = __builtin_amdgcn_wmma_f32_16x16x32_bf16(
                    false, aF[mi], false, bF[ni], (short)0, acc[mi][ni], false, false);
    }

    // C/D layout: M = r + 8*(lane>>4), N = lane&15 within each 16x16 tile.
#pragma unroll
    for (int mi = 0; mi < 2; ++mi) {
#pragma unroll
        for (int ni = 0; ni < 4; ++ni) {
            const int   nCol = nBase + ni * 16 + laneRow;
            const float bN   = bias[nCol];
#pragma unroll
            for (int r = 0; r < 8; ++r) {
                const int    mRow = mBase + mi * 16 + laneKOff + r;
                const float  val  = acc[mi][ni][r] + bN;
                const size_t idx  = (size_t)mRow * N + nCol;
                if (STORE_BF16) ((__bf16*)Out)[idx] = (__bf16)val;
                else            ((float*)Out)[idx]  = val;
            }
        }
    }
}

// ---------------------------------------------------------------------------
// Kernel 2: per token t: S = softmax((Q_t K_t^T)/sqrt(32)); out = S * V_t.
// One wave per token; scores via 2x2 WMMA tiles; then LDS transpose so lane h
// owns score row h -> softmax entirely in registers (no shuffles), then
// out[h,:] = S[h,:] * V[h,:] with b128 vector loads/stores.
// ---------------------------------------------------------------------------
__global__ __launch_bounds__(256) void attn_combine_kernel(
    const __bf16* __restrict__ Q, const __bf16* __restrict__ Kb,
    const float* __restrict__ V, float* __restrict__ Out, int nTok)
{
    __shared__ float sc[8][32][33];   // per-wave 32x32 tile, stride 33: conflict-free

    const int lane = threadIdx.x & 31;
    const int wave = threadIdx.x >> 5;
    const int tok  = blockIdx.x * 8 + wave;
    if (tok >= nTok) return;          // wave-uniform; EXEC all-ones for WMMA

    const __bf16* qt = Q  + (size_t)tok * 1024;
    const __bf16* kt = Kb + (size_t)tok * 1024;
    const int laneRow  = lane & 15;
    const int laneKOff = (lane >> 4) << 3;

    bf16x16 qF[2], kF[2];
#pragma unroll
    for (int t = 0; t < 2; ++t) {
        qF[t] = load_frag_bf16(qt + (size_t)(t * 16 + laneRow) * 32 + laneKOff);
        kF[t] = load_frag_bf16(kt + (size_t)(t * 16 + laneRow) * 32 + laneKOff);
    }

    f32x8 s[2][2] = {};
#pragma unroll
    for (int mi = 0; mi < 2; ++mi)
#pragma unroll
        for (int ni = 0; ni < 2; ++ni)
            s[mi][ni] = __builtin_amdgcn_wmma_f32_16x16x32_bf16(
                false, qF[mi], false, kF[ni], (short)0, s[mi][ni], false, false);

    const float scale = 0.17677669529663687f;  // 1/sqrt(32)

    // Scatter scores to LDS: element (h = mi*16 + laneKOff + r, d = ni*16 + laneRow)
#pragma unroll
    for (int mi = 0; mi < 2; ++mi)
#pragma unroll
        for (int ni = 0; ni < 2; ++ni)
#pragma unroll
            for (int r = 0; r < 8; ++r)
                sc[wave][mi * 16 + laneKOff + r][ni * 16 + laneRow] = s[mi][ni][r] * scale;

    // Intra-wave LDS ops are in-order; wait + compiler barrier before reading.
    asm volatile("s_wait_dscnt 0x0" ::: "memory");

    // Lane h = lane owns score row h (stride-33 rows: bank-conflict-free).
    float row[32];
#pragma unroll
    for (int j = 0; j < 32; ++j) row[j] = sc[wave][lane][j];

    float mx = row[0];
#pragma unroll
    for (int j = 1; j < 32; ++j) mx = fmaxf(mx, row[j]);
    float sum = 0.0f;
#pragma unroll
    for (int j = 0; j < 32; ++j) { row[j] = __expf(row[j] - mx); sum += row[j]; }
    const float inv = 1.0f / sum;

    // out[h,d] = S[h,d] * V[h,d]  (nh == hd elementwise combine)
    const float* vt = V   + (size_t)tok * 1024 + lane * 32;
    float*       ot = Out + (size_t)tok * 1024 + lane * 32;
#pragma unroll
    for (int j4 = 0; j4 < 8; ++j4) {
        f32x4 vv = *(const f32x4*)(vt + j4 * 4);
        f32x4 oo;
#pragma unroll
        for (int i = 0; i < 4; ++i) oo[i] = row[j4 * 4 + i] * inv * vv[i];
        *(f32x4*)(ot + j4 * 4) = oo;
    }
}

// ---------------------------------------------------------------------------
// Launcher
// ---------------------------------------------------------------------------
extern "C" void kernel_launch(void* const* d_in, const int* in_sizes, int n_in,
                              void* d_out, int out_size, void* d_ws, size_t ws_size,
                              hipStream_t stream) {
    const float* x  = (const float*)d_in[0];
    const float* Wq = (const float*)d_in[1];
    const float* bq = (const float*)d_in[2];
    const float* Wk = (const float*)d_in[3];
    const float* bk = (const float*)d_in[4];
    const float* Wv = (const float*)d_in[5];
    const float* bv = (const float*)d_in[6];

    const int M = 4 * 4096;   // B*S tokens
    const int K = 1024;
    const int N = 1024;

    // Workspace layout (bytes): xb 32M | Wq/Wk/Wv bf16 2M each | Q 32M | K 32M | V 64M
    __bf16* xb  = (__bf16*)d_ws;
    __bf16* wqb = xb  + (size_t)M * K;
    __bf16* wkb = wqb + (size_t)N * K;
    __bf16* wvb = wkb + (size_t)N * K;
    __bf16* qws = wvb + (size_t)N * K;
    __bf16* kws = qws + (size_t)M * N;
    float*  vws = (float*)(kws + (size_t)M * N);

    // 0) one-time bf16 packs (grids divide exactly: 8 elems/thread, 256 thr/blk)
    pack_bf16_kernel<<<(unsigned)((size_t)M * K / 2048), 256, 0, stream>>>(x,  xb,  (size_t)M * K);
    pack_bf16_kernel<<<(unsigned)((size_t)N * K / 2048), 256, 0, stream>>>(Wq, wqb, (size_t)N * K);
    pack_bf16_kernel<<<(unsigned)((size_t)N * K / 2048), 256, 0, stream>>>(Wk, wkb, (size_t)N * K);
    pack_bf16_kernel<<<(unsigned)((size_t)N * K / 2048), 256, 0, stream>>>(Wv, wvb, (size_t)N * K);

    // 1) QKV projections
    dim3 grid(M / 64, N / 256);   // 256 x 4 workgroups
    qkv_gemm_kernel<true ><<<grid, 256, 0, stream>>>(xb, wqb, bq, (void*)qws, M, K, N);
    qkv_gemm_kernel<true ><<<grid, 256, 0, stream>>>(xb, wkb, bk, (void*)kws, M, K, N);
    qkv_gemm_kernel<false><<<grid, 256, 0, stream>>>(xb, wvb, bv, (void*)vws, M, K, N);

    // 2) per-token scores + softmax + elementwise V
    attn_combine_kernel<<<M / 8, 256, 0, stream>>>(qws, kws, vws, (float*)d_out, M);
}
